// MultiHeadSelfAttention_1434519077425
// MI455X (gfx1250) — matrix operand order repaired and checked by reading, not compiled, not run on hardware
//
#include <hip/hip_runtime.h>
#include <hip/hip_bf16.h>

// ---------------------------------------------------------------------------
// MHA forward for MI455X (gfx1250, wave32, WMMA).
//   B=2, S=2048, D=1024, H=16, HD=64, causal, softmax scale = 1/8.
// Pipeline: f32->f16 staging, QKV GEMM (WMMA, 32x64 wave tiles), flash
// attention (WMMA + LDS P-transpose + online softmax), projection GEMM.
// Workspace requirement: ~48 MB (assumed available in d_ws).
// ---------------------------------------------------------------------------

#define BATCH 2
#define SEQ   2048
#define DM    1024
#define NH    16
#define HD    64
#define QSCALE 0.125f   // 1/sqrt(64)

typedef __attribute__((ext_vector_type(16))) _Float16 v16h;
typedef __attribute__((ext_vector_type(8)))  _Float16 v8h;
typedef __attribute__((ext_vector_type(8)))  float    v8f;

// A-matrix 16x32 f16 fragment: lane row = L%16, K = 16*(L>=16) + 0..15
// (contiguous, matching B). p = row + koff + kb with kb = 8*(L>=16),
// => vector = 16 halves @ p + kb.
__device__ __forceinline__ v16h make_afrag(const _Float16* p) {
  return *(const v16h*)(p + ((threadIdx.x & 31) >> 4) * 8);
}

// B-matrix 32x16 f16 fragment: lane col = L%16; lanes 0-15 hold K=0..15,
// lanes 16-31 hold K=16..31 => one contiguous 16-half (32B) chunk per lane.
__device__ __forceinline__ v16h load_bfrag(const _Float16* p) {
  return *(const v16h*)p;
}

__device__ __forceinline__ v8f wmma_f16(v16h a, v16h b, v8f c) {
  return __builtin_amdgcn_wmma_f32_16x16x32_f16(
      /*neg_a=*/false, a, /*neg_b=*/false, b,
      /*c_mod=*/(short)0, c, /*reuse_a=*/false, /*reuse_b=*/false);
}

// ------------------------------- prep kernels -------------------------------

__global__ __launch_bounds__(256) void cvt_f32_to_f16(
    const float* __restrict__ in, _Float16* __restrict__ out, int n) {
  int i = blockIdx.x * blockDim.x + threadIdx.x;
  if (i < n) out[i] = (_Float16)in[i];
}

// Wt[n*K + k] = (f16) W[k*N + n]   (K-major for contiguous B fragments)
__global__ __launch_bounds__(256) void transpose_w_f16(
    const float* __restrict__ W, _Float16* __restrict__ Wt, int K, int N) {
  int i = blockIdx.x * blockDim.x + threadIdx.x;
  if (i < N * K) {
    int n = i / K, k = i % K;
    Wt[i] = (_Float16)W[(size_t)k * N + n];
  }
}

// ------------------------------- QKV GEMM ----------------------------------
// xh[4096,1024] (f16) @ Wqkvt[3072,1024]^T + b -> scatter Q/K/Vt (f16).
// One wave computes a 32(M) x 64(N) tile: 2 A frags, 4 B frags, 8 WMMAs per
// K-step of 32 -> B fragments amortized over two M sub-tiles.

__global__ __launch_bounds__(128) void qkv_gemm(
    const _Float16* __restrict__ xh, const _Float16* __restrict__ Wt,
    const float* __restrict__ bias, _Float16* __restrict__ Qh,
    _Float16* __restrict__ Kh, _Float16* __restrict__ Vth) {
  const int wave = threadIdx.x >> 5;
  const int lane = threadIdx.x & 31;
  const int lcol = lane & 15;
  const int hlf  = lane >> 4;
  const int kb   = hlf * 8;

  const int wl = blockIdx.x * 4 + wave;   // 6144 waves total
  const int m0 = (wl / 48) * 32;          // 128 M tiles (32 rows)
  const int n0 = (wl % 48) * 64;          // 48 N64 tiles

  v8f c[2][4] = {};
  const _Float16* arow0 = xh + (size_t)(m0 + lcol) * DM;
  const _Float16* arow1 = xh + (size_t)(m0 + 16 + lcol) * DM;
  const _Float16* b0r = Wt + (size_t)(n0 + 0 * 16 + lcol) * DM + hlf * 16;
  const _Float16* b1r = Wt + (size_t)(n0 + 1 * 16 + lcol) * DM + hlf * 16;
  const _Float16* b2r = Wt + (size_t)(n0 + 2 * 16 + lcol) * DM + hlf * 16;
  const _Float16* b3r = Wt + (size_t)(n0 + 3 * 16 + lcol) * DM + hlf * 16;

  for (int kk = 0; kk < DM; kk += 32) {
    const v16h a0 = make_afrag(arow0 + kk + kb);
    const v16h a1 = make_afrag(arow1 + kk + kb);
    const v16h bf0 = load_bfrag(b0r + kk);
    const v16h bf1 = load_bfrag(b1r + kk);
    const v16h bf2 = load_bfrag(b2r + kk);
    const v16h bf3 = load_bfrag(b3r + kk);
    c[0][0] = wmma_f16(a0, bf0, c[0][0]);
    c[1][0] = wmma_f16(a1, bf0, c[1][0]);
    c[0][1] = wmma_f16(a0, bf1, c[0][1]);
    c[1][1] = wmma_f16(a1, bf1, c[1][1]);
    c[0][2] = wmma_f16(a0, bf2, c[0][2]);
    c[1][2] = wmma_f16(a1, bf2, c[1][2]);
    c[0][3] = wmma_f16(a0, bf3, c[0][3]);
    c[1][3] = wmma_f16(a1, bf3, c[1][3]);
  }

  // Epilogue: n0 is a multiple of 64 -> (which, head) constant for the tile.
  const int which = n0 / DM;             // 0=Q 1=K 2=V
  const int hh    = (n0 % DM) / HD;
#pragma unroll
  for (int a = 0; a < 2; ++a) {
#pragma unroll
    for (int f = 0; f < 4; ++f) {
      const int d = f * 16 + lcol;       // 0..63
      const float bv = bias[n0 + f * 16 + lcol];
#pragma unroll
      for (int r = 0; r < 8; ++r) {
        const int m = m0 + a * 16 + r + 8 * hlf;
        const int bb = m >> 11;          // m / SEQ
        const int s  = m & (SEQ - 1);
        const float v = c[a][f][r] + bv;
        const size_t bh = (size_t)(bb * NH + hh);
        if (which == 0)
          Qh[(bh * SEQ + s) * HD + d] = (_Float16)(v * QSCALE);
        else if (which == 1)
          Kh[(bh * SEQ + s) * HD + d] = (_Float16)v;
        else
          Vth[(bh * HD + d) * SEQ + s] = (_Float16)v;  // V^T [B,H,HD,S]
      }
    }
  }
}

// ---------------------------- flash attention ------------------------------
// One wave per 16-query tile; key tiles of 32; online softmax; causal.

__global__ __launch_bounds__(128) void attn_fused(
    const _Float16* __restrict__ Qh, const _Float16* __restrict__ Kh,
    const _Float16* __restrict__ Vth, _Float16* __restrict__ attnh) {
  __shared__ _Float16 Pt[4][16 * 32];    // per-wave P staging (1 KB each)
  const int wave = threadIdx.x >> 5;
  const int lane = threadIdx.x & 31;
  const int lcol = lane & 15;
  const int hlf  = lane >> 4;
  const int kb   = hlf * 8;

  const int q0 = (blockIdx.x * 4 + wave) * 16;   // query tile base
  const int bh = blockIdx.y;                     // 0..31
  const int b  = bh >> 4;
  const int h  = bh & 15;

  const _Float16* Qb = Qh + (size_t)bh * SEQ * HD;
  const _Float16* Kb = Kh + (size_t)bh * SEQ * HD;
  const _Float16* Vb = Vth + (size_t)bh * HD * SEQ;

  // Q fragments (16x32 A-matrix), koff = 0 and 32. Q already holds 1/sqrt(d).
  const _Float16* qrow = Qb + (size_t)(q0 + lcol) * HD;
  const v16h qf0 = make_afrag(qrow + 0 + kb);
  const v16h qf1 = make_afrag(qrow + 32 + kb);

  v8f o0 = {}, o1 = {}, o2 = {}, o3 = {};
  float m_run[8], l_run[8];
#pragma unroll
  for (int r = 0; r < 8; ++r) { m_run[r] = -3.0e38f; l_run[r] = 0.0f; }

  _Float16* pt = &Pt[wave][0];
  const int nkt = (q0 + 16 + 31) >> 5;   // causal key-tile bound

  for (int kt = 0; kt < nkt; ++kt) {
    const int k0 = kt << 5;

    // ---- scores: two 16x16 C tiles over 32 keys (4 WMMAs) ----
    v8f s0 = {}, s1 = {};
    {
      const _Float16* kr0 = Kb + (size_t)(k0 + lcol) * HD + hlf * 16;
      s0 = wmma_f16(qf0, load_bfrag(kr0 + 0), s0);
      s0 = wmma_f16(qf1, load_bfrag(kr0 + 32), s0);
      const _Float16* kr1 = Kb + (size_t)(k0 + 16 + lcol) * HD + hlf * 16;
      s1 = wmma_f16(qf0, load_bfrag(kr1 + 0), s1);
      s1 = wmma_f16(qf1, load_bfrag(kr1 + 32), s1);
    }

    // ---- causal mask (only the boundary tiles need it) ----
    if (k0 + 31 > q0) {
#pragma unroll
      for (int r = 0; r < 8; ++r) {
        const int qq = q0 + r + 8 * hlf;
        if (k0 + lcol > qq)      s0[r] = -3.0e38f;
        if (k0 + 16 + lcol > qq) s1[r] = -3.0e38f;
      }
    }

    // ---- online softmax: row reductions across the 16-lane half ----
    float sc[8], rs[8];
#pragma unroll
    for (int r = 0; r < 8; ++r) {
      float mt = fmaxf(s0[r], s1[r]);
      mt = fmaxf(mt, __shfl_xor(mt, 8));
      mt = fmaxf(mt, __shfl_xor(mt, 4));
      mt = fmaxf(mt, __shfl_xor(mt, 2));
      mt = fmaxf(mt, __shfl_xor(mt, 1));
      const float mn = fmaxf(m_run[r], mt);
      sc[r] = __expf(m_run[r] - mn);
      m_run[r] = mn;
      const float p0 = __expf(s0[r] - mn);
      const float p1 = __expf(s1[r] - mn);
      s0[r] = p0; s1[r] = p1;
      float t = p0 + p1;
      t += __shfl_xor(t, 8);
      t += __shfl_xor(t, 4);
      t += __shfl_xor(t, 2);
      t += __shfl_xor(t, 1);
      rs[r] = t;
    }
#pragma unroll
    for (int r = 0; r < 8; ++r) {
      l_run[r] = l_run[r] * sc[r] + rs[r];
      o0[r] *= sc[r]; o1[r] *= sc[r]; o2[r] *= sc[r]; o3[r] *= sc[r];
    }

    // ---- C->A fragment transpose of P through LDS (per-wave, no barrier) ----
#pragma unroll
    for (int r = 0; r < 8; ++r) {
      const int row = r + 8 * hlf;
      pt[row * 32 + lcol]      = (_Float16)s0[r];
      pt[row * 32 + 16 + lcol] = (_Float16)s1[r];
    }
    asm volatile("s_wait_dscnt 0x0" ::: "memory");
    const v16h pf = make_afrag(pt + lcol * 32 + kb);

    // ---- P(16x32) @ V(32x64): 4 WMMAs; V^T layout gives contiguous B frags --
    const _Float16* vb = Vb + (size_t)lcol * SEQ + k0 + hlf * 16;
    o0 = wmma_f16(pf, load_bfrag(vb + (size_t)0 * 16 * SEQ), o0);
    o1 = wmma_f16(pf, load_bfrag(vb + (size_t)1 * 16 * SEQ), o1);
    o2 = wmma_f16(pf, load_bfrag(vb + (size_t)2 * 16 * SEQ), o2);
    o3 = wmma_f16(pf, load_bfrag(vb + (size_t)3 * 16 * SEQ), o3);
  }

  // ---- finalize: O /= l, store f16 into [B,S,H,HD] for the proj GEMM ----
#pragma unroll
  for (int r = 0; r < 8; ++r) {
    const int q = q0 + r + 8 * hlf;
    const float inv = 1.0f / l_run[r];
    _Float16* op = attnh + ((size_t)(b * SEQ + q) * NH + h) * HD + lcol;
    op[0]  = (_Float16)(o0[r] * inv);
    op[16] = (_Float16)(o1[r] * inv);
    op[32] = (_Float16)(o2[r] * inv);
    op[48] = (_Float16)(o3[r] * inv);
  }
}

// ----------------------------- projection GEMM -----------------------------
// attnh[4096,1024] (f16) @ Wprojt[1024,1024]^T + b -> out f32 [4096,1024].
// Same 32x64 wave-tile engine as qkv_gemm.

__global__ __launch_bounds__(128) void proj_gemm(
    const _Float16* __restrict__ ah, const _Float16* __restrict__ Wt,
    const float* __restrict__ bias, float* __restrict__ out) {
  const int wave = threadIdx.x >> 5;
  const int lane = threadIdx.x & 31;
  const int lcol = lane & 15;
  const int hlf  = lane >> 4;
  const int kb   = hlf * 8;

  const int wl = blockIdx.x * 4 + wave;   // 2048 waves total
  const int m0 = (wl / 16) * 32;          // 128 M tiles (32 rows)
  const int n0 = (wl % 16) * 64;          // 16 N64 tiles

  v8f c[2][4] = {};
  const _Float16* arow0 = ah + (size_t)(m0 + lcol) * DM;
  const _Float16* arow1 = ah + (size_t)(m0 + 16 + lcol) * DM;
  const _Float16* b0r = Wt + (size_t)(n0 + 0 * 16 + lcol) * DM + hlf * 16;
  const _Float16* b1r = Wt + (size_t)(n0 + 1 * 16 + lcol) * DM + hlf * 16;
  const _Float16* b2r = Wt + (size_t)(n0 + 2 * 16 + lcol) * DM + hlf * 16;
  const _Float16* b3r = Wt + (size_t)(n0 + 3 * 16 + lcol) * DM + hlf * 16;

  for (int kk = 0; kk < DM; kk += 32) {
    const v16h a0 = make_afrag(arow0 + kk + kb);
    const v16h a1 = make_afrag(arow1 + kk + kb);
    const v16h bf0 = load_bfrag(b0r + kk);
    const v16h bf1 = load_bfrag(b1r + kk);
    const v16h bf2 = load_bfrag(b2r + kk);
    const v16h bf3 = load_bfrag(b3r + kk);
    c[0][0] = wmma_f16(a0, bf0, c[0][0]);
    c[1][0] = wmma_f16(a1, bf0, c[1][0]);
    c[0][1] = wmma_f16(a0, bf1, c[0][1]);
    c[1][1] = wmma_f16(a1, bf1, c[1][1]);
    c[0][2] = wmma_f16(a0, bf2, c[0][2]);
    c[1][2] = wmma_f16(a1, bf2, c[1][2]);
    c[0][3] = wmma_f16(a0, bf3, c[0][3]);
    c[1][3] = wmma_f16(a1, bf3, c[1][3]);
  }

#pragma unroll
  for (int a = 0; a < 2; ++a) {
#pragma unroll
    for (int f = 0; f < 4; ++f) {
      const int n = n0 + f * 16 + lcol;
      const float bv = bias[n];
#pragma unroll
      for (int r = 0; r < 8; ++r) {
        const int m = m0 + a * 16 + r + 8 * hlf;
        out[(size_t)m * DM + n] = c[a][f][r] + bv;
      }
    }
  }
}

// ------------------------------- launcher ----------------------------------

extern "C" void kernel_launch(void* const* d_in, const int* in_sizes, int n_in,
                              void* d_out, int out_size, void* d_ws, size_t ws_size,
                              hipStream_t stream) {
  const float* x     = (const float*)d_in[0];   // [2,2048,1024]
  const float* Wqkv  = (const float*)d_in[1];   // [1024,3072]
  const float* bqkv  = (const float*)d_in[2];   // [3072]
  const float* Wproj = (const float*)d_in[3];   // [1024,1024]
  const float* bproj = (const float*)d_in[4];   // [1024]
  float* out = (float*)d_out;                   // [2,2048,1024] f32

  // Workspace carve-up (256B aligned). Total ~48 MB.
  size_t off = 0;
  auto carve = [&](size_t bytes) {
    void* p = (char*)d_ws + off;
    off += (bytes + 255) & ~(size_t)255;
    return p;
  };
  const size_t M = (size_t)BATCH * SEQ;              // 4096
  _Float16* xh     = (_Float16*)carve(M * DM * 2);             // 8 MB
  _Float16* Wqkvt  = (_Float16*)carve((size_t)3 * DM * DM * 2); // 6 MB
  _Float16* Wprojt = (_Float16*)carve((size_t)DM * DM * 2);     // 2 MB
  _Float16* Qh     = (_Float16*)carve(M * DM * 2);             // 8 MB
  _Float16* Kh     = (_Float16*)carve(M * DM * 2);             // 8 MB
  _Float16* Vth    = (_Float16*)carve(M * DM * 2);             // 8 MB
  _Float16* attnh  = (_Float16*)carve(M * DM * 2);             // 8 MB

  // 1) staging: f32 -> f16 (x), transpose+convert weights to K-major f16
  {
    int n = (int)(M * DM);
    cvt_f32_to_f16<<<n / 256, 256, 0, stream>>>(x, xh, n);
  }
  {
    int n = 3 * DM * DM;
    transpose_w_f16<<<n / 256, 256, 0, stream>>>(Wqkv, Wqkvt, DM, 3 * DM);
  }
  {
    int n = DM * DM;
    transpose_w_f16<<<n / 256, 256, 0, stream>>>(Wproj, Wprojt, DM, DM);
  }

  // 2) QKV GEMM: 128 x 48 wave-tiles (32x64 each), 4 waves/block -> 1536 blocks
  qkv_gemm<<<1536, 128, 0, stream>>>(xh, Wqkvt, bqkv, Qh, Kh, Vth);

  // 3) flash attention: grid (32 q-blocks, 32 bh), 4 waves/block
  attn_fused<<<dim3(32, 32), 128, 0, stream>>>(Qh, Kh, Vth, attnh);

  // 4) projection GEMM: 128 x 16 wave-tiles -> 512 blocks
  proj_gemm<<<512, 128, 0, stream>>>(attnh, Wprojt, bproj, out);
}
